// HyperGATConv_47004122087477
// MI455X (gfx1250) — matrix operand order repaired and checked by reading, not compiled, not run on hardware
//
#include <hip/hip_runtime.h>
#include <hip/hip_bf16.h>
#include <math.h>

typedef __attribute__((ext_vector_type(16))) _Float16 v16h;
typedef __attribute__((ext_vector_type(8)))  float    v8f;

#define IN_DIM    128
#define OUT_DIM   64
#define NUM_HEADS 8
#define HEAD_DIM  8
#define ALPHA     0.2f
#define BCOLS     80      // 64 weighted cols + 8 expn cols + 8 zero pad (5 x 16 tiles)
#define KTILE     32
#define KPART     800     // K elements per partition = 25 wmma steps

__device__ __forceinline__ unsigned enc_f32(float f) {
  unsigned u = __float_as_uint(f);
  return (u & 0x80000000u) ? ~u : (u | 0x80000000u);
}
__device__ __forceinline__ float dec_f32(unsigned u) {
  return (u & 0x80000000u) ? __uint_as_float(u ^ 0x80000000u) : __uint_as_float(~u);
}

// out[n,j] = bias[j]; nd zeroed; smax zeroed (enc of -inf-ish)
__global__ void init_kernel(float* __restrict__ out, float* __restrict__ nd,
                            unsigned* __restrict__ smax, const float* __restrict__ bias,
                            int N, int E) {
  int tid = blockIdx.x * blockDim.x + threadIdx.x;
  if (tid < N * OUT_DIM) out[tid] = bias[tid & (OUT_DIM - 1)];
  if (tid < E * BCOLS)   nd[tid] = 0.0f;
  if (tid < NUM_HEADS)   smax[tid] = 0u;
}

// Xh = X @ W  (f32 -> f16 operands, f32 accumulate). One wave per 16 rows.
__global__ __launch_bounds__(32) void xh_gemm_kernel(const float* __restrict__ X,
                                                     const float* __restrict__ W,
                                                     float* __restrict__ Xh, int N) {
  const int lane = threadIdx.x & 31;
  const int g = lane >> 4, m = lane & 15;
  const int n0 = blockIdx.x * 16;
  v8f c[4] = {};
#pragma unroll
  for (int kb = 0; kb < IN_DIM; kb += KTILE) {
    v16h a;
    const float* ap = X + (size_t)(n0 + m) * IN_DIM + kb + g * 16;
#pragma unroll
    for (int i = 0; i < 8; ++i) {
      a[i]     = (_Float16)ap[i];
      a[8 + i] = (_Float16)ap[8 + i];
    }
#pragma unroll
    for (int t = 0; t < 4; ++t) {
      v16h b;
      const float* bp = W + (size_t)(kb + g * 16) * OUT_DIM + t * 16 + m;
#pragma unroll
      for (int e = 0; e < 16; ++e) b[e] = (_Float16)bp[(size_t)e * OUT_DIM];
      c[t] = __builtin_amdgcn_wmma_f32_16x16x32_f16(false, a, false, b, (short)0, c[t],
                                                    false, false);
    }
  }
#pragma unroll
  for (int t = 0; t < 4; ++t)
#pragma unroll
    for (int v = 0; v < 8; ++v)
      Xh[(size_t)(n0 + v + 8 * g) * OUT_DIM + t * 16 + m] = c[t][v];
}

// s[n,h] = leaky_relu(Xh[n,h,:] . (att[:,:d] + att[:,d:])); global per-head max via atomicMax
__global__ __launch_bounds__(256) void score_kernel(const float* __restrict__ Xh,
                                                    const float* __restrict__ att,
                                                    float* __restrict__ s,
                                                    unsigned* __restrict__ smax, int N) {
  __shared__ float red[256];
  const int t = threadIdx.x;
  const int tid = blockIdx.x * 256 + t;
  const int h = tid & (NUM_HEADS - 1);
  const int n = tid >> 3;
  float sv = -INFINITY;
  if (n < N) {
    float acc = 0.f;
#pragma unroll
    for (int d = 0; d < HEAD_DIM; ++d) {
      float ah = att[h * 2 * HEAD_DIM + d] + att[h * 2 * HEAD_DIM + HEAD_DIM + d];
      acc += ah * Xh[(size_t)n * OUT_DIM + h * HEAD_DIM + d];
    }
    sv = (acc > 0.f) ? acc : ALPHA * acc;
    s[(size_t)n * NUM_HEADS + h] = sv;
  }
  red[t] = sv;
  __syncthreads();
  for (int off = 128; off >= 8; off >>= 1) {  // offsets are multiples of 8: same-head reduce
    if (t < off) red[t] = fmaxf(red[t], red[t + off]);
    __syncthreads();
  }
  if (t < NUM_HEADS) atomicMax(&smax[t], enc_f32(red[t]));
}

// B1T[c, n] (f16, transposed): c<64 -> expn*Xh ; c in [64,72) -> expn ; c in [72,80) -> 0
__global__ void bmat_kernel(const float* __restrict__ Xh, const float* __restrict__ s,
                            const unsigned* __restrict__ smax, _Float16* __restrict__ B1T,
                            int N) {
  int tid = blockIdx.x * blockDim.x + threadIdx.x;
  if (tid >= BCOLS * N) return;
  int c = tid / N, n = tid - c * N;
  _Float16 val = (_Float16)0.0f;
  if (c < OUT_DIM) {
    int h = c >> 3;
    float e = __expf(s[(size_t)n * NUM_HEADS + h] - dec_f32(smax[h]));
    val = (_Float16)(e * Xh[(size_t)n * OUT_DIM + c]);
  } else if (c < OUT_DIM + NUM_HEADS) {
    int h = c - OUT_DIM;
    val = (_Float16)__expf(s[(size_t)n * NUM_HEADS + h] - dec_f32(smax[h]));
  }
  B1T[tid] = val;
}

// nd[E,80] += maskT(16e x 800n part) @ B1T : mask built on the fly from H (column reads)
__global__ __launch_bounds__(32) void edge_gemm_kernel(const int* __restrict__ H,
                                                       const _Float16* __restrict__ B1T,
                                                       float* __restrict__ nd, int N, int E) {
  const int lane = threadIdx.x & 31;
  const int g = lane >> 4, m = lane & 15;
  const int e0 = blockIdx.x * 16;
  const int kbase = blockIdx.y * KPART;
  const _Float16 one = (_Float16)1.0f, zero = (_Float16)0.0f;
  v8f c[5] = {};
  for (int it = 0; it < KPART / KTILE; ++it) {
    const int kb = kbase + it * KTILE;
    v16h a;
    const int* hp = H + (size_t)(kb + g * 16) * E + e0 + m;
#pragma unroll
    for (int i = 0; i < 8; ++i) {
      a[i]     = hp[(size_t)i * E]        ? one : zero;
      a[8 + i] = hp[(size_t)(i + 8) * E] ? one : zero;
    }
#pragma unroll
    for (int t = 0; t < 5; ++t) {
      const v16h b = *(const v16h*)(B1T + (size_t)(t * 16 + m) * N + kb + g * 16);
      c[t] = __builtin_amdgcn_wmma_f32_16x16x32_f16(false, a, false, b, (short)0, c[t],
                                                    false, false);
    }
  }
#pragma unroll
  for (int t = 0; t < 5; ++t)
#pragma unroll
    for (int v = 0; v < 8; ++v)
      atomicAdd(&nd[(size_t)(e0 + v + 8 * g) * BCOLS + t * 16 + m], c[t][v]);
}

// aggT[c, e] (f16, transposed) = denom>0 ? num/denom : 0
__global__ void agg_kernel(const float* __restrict__ nd, _Float16* __restrict__ aggT, int E) {
  int tid = blockIdx.x * blockDim.x + threadIdx.x;
  if (tid >= OUT_DIM * E) return;
  int c = tid / E, e = tid - c * E;
  int h = c >> 3;
  float den = nd[(size_t)e * BCOLS + OUT_DIM + h];
  float num = nd[(size_t)e * BCOLS + c];
  aggT[tid] = (_Float16)(den > 0.f ? num / den : 0.f);
}

// out[N,64] += mask(16n x 800e part) @ agg : contiguous H row reads
__global__ __launch_bounds__(32) void node_gemm_kernel(const int* __restrict__ H,
                                                       const _Float16* __restrict__ aggT,
                                                       float* __restrict__ out, int E) {
  const int lane = threadIdx.x & 31;
  const int g = lane >> 4, m = lane & 15;
  const int n0 = blockIdx.x * 16;
  const int kbase = blockIdx.y * KPART;
  const _Float16 one = (_Float16)1.0f, zero = (_Float16)0.0f;
  v8f c[4] = {};
  for (int it = 0; it < KPART / KTILE; ++it) {
    const int kb = kbase + it * KTILE;
    v16h a;
    const int* hp = H + (size_t)(n0 + m) * E + kb + g * 16;
#pragma unroll
    for (int i = 0; i < 8; ++i) {
      a[i]     = hp[i]      ? one : zero;
      a[8 + i] = hp[8 + i] ? one : zero;
    }
#pragma unroll
    for (int t = 0; t < 4; ++t) {
      const v16h b = *(const v16h*)(aggT + (size_t)(t * 16 + m) * E + kb + g * 16);
      c[t] = __builtin_amdgcn_wmma_f32_16x16x32_f16(false, a, false, b, (short)0, c[t],
                                                    false, false);
    }
  }
#pragma unroll
  for (int t = 0; t < 4; ++t)
#pragma unroll
    for (int v = 0; v < 8; ++v)
      atomicAdd(&out[(size_t)(n0 + v + 8 * g) * OUT_DIM + t * 16 + m], c[t][v]);
}

extern "C" void kernel_launch(void* const* d_in, const int* in_sizes, int n_in,
                              void* d_out, int out_size, void* d_ws, size_t ws_size,
                              hipStream_t stream) {
  const float* X    = (const float*)d_in[0];
  const int*   H    = (const int*)d_in[1];
  const float* W    = (const float*)d_in[2];
  const float* att  = (const float*)d_in[3];
  const float* bias = (const float*)d_in[4];
  float* out = (float*)d_out;

  const int N = in_sizes[0] / IN_DIM;   // 20000
  const int E = in_sizes[1] / N;        // 4000
  (void)n_in; (void)out_size; (void)ws_size;

  char* ws = (char*)d_ws;
  size_t off = 0;
  auto carve = [&](size_t bytes) -> void* {
    void* p = ws + off;
    off = (off + bytes + 255) & ~(size_t)255;
    return p;
  };
  float*    Xh   = (float*)carve((size_t)N * OUT_DIM * sizeof(float));
  float*    s    = (float*)carve((size_t)N * NUM_HEADS * sizeof(float));
  unsigned* smax = (unsigned*)carve(256);
  _Float16* B1T  = (_Float16*)carve((size_t)BCOLS * N * sizeof(_Float16));
  float*    nd   = (float*)carve((size_t)E * BCOLS * sizeof(float));
  _Float16* aggT = (_Float16*)carve((size_t)OUT_DIM * E * sizeof(_Float16));

  int initN = (N * OUT_DIM > E * BCOLS) ? N * OUT_DIM : E * BCOLS;
  init_kernel<<<(initN + 255) / 256, 256, 0, stream>>>(out, nd, smax, bias, N, E);

  xh_gemm_kernel<<<N / 16, 32, 0, stream>>>(X, W, Xh, N);

  score_kernel<<<(N * NUM_HEADS + 255) / 256, 256, 0, stream>>>(Xh, att, s, smax, N);

  bmat_kernel<<<(BCOLS * N + 255) / 256, 256, 0, stream>>>(Xh, s, smax, B1T, N);

  edge_gemm_kernel<<<dim3(E / 16, N / KPART), 32, 0, stream>>>(H, B1T, nd, N, E);

  agg_kernel<<<(OUT_DIM * E + 255) / 256, 256, 0, stream>>>(nd, aggT, E);

  node_gemm_kernel<<<dim3(N / 16, E / KPART), 32, 0, stream>>>(H, aggT, out, E);
}
